// MultiAgentSEPSNetworkParallel_1846835937333
// MI455X (gfx1250) — matrix operand order repaired and checked by reading, not compiled, not run on hardware
//
#include <hip/hip_runtime.h>
#include <hip/hip_bf16.h>

// ---------------- problem constants (match reference) ----------------
#define A_ 8
#define B_ 2048
#define D_ 128
#define N_ 4
#define H1_ 1024
#define H2_ 1024
#define H3_ 512
#define ROWS_ (A_ * B_)                 // 16384
#define MPAD_ 32                        // bucket padding / per-wave M tile
#define MAXSLOTS_ (ROWS_ + N_ * MPAD_)  // 16512
#define MTILES_ (MAXSLOTS_ / MPAD_)     // 516 blocks in M

typedef __attribute__((ext_vector_type(16))) _Float16 v16h;
typedef __attribute__((ext_vector_type(8)))  float    v8f;

union Frag16 { v16h v; int4 q[2]; };

// ---------------- bucketing kernels ----------------
__global__ void seps_init_ws(int* cnt, int* perm, int nperm) {
    int i = blockIdx.x * blockDim.x + threadIdx.x;
    if (i < 4) cnt[i] = 0;
    if (i < nperm) perm[i] = -1;
}

__global__ void seps_hist(const int* __restrict__ seps, int* __restrict__ cnt) {
    int r = blockIdx.x * blockDim.x + threadIdx.x;
    if (r >= ROWS_) return;
    int a = r / B_, b = r % B_;
    int net = seps[b * A_ + a];          // seps_indices is [B, A]
    atomicAdd(&cnt[net], 1);
}

__global__ void seps_offsets(const int* __restrict__ cnt, int* __restrict__ start,
                             int* __restrict__ cursor) {
    if (threadIdx.x == 0 && blockIdx.x == 0) {
        int s = 0;
        for (int n = 0; n < N_; ++n) {
            start[n] = s;
            cursor[n] = s;
            s += ((cnt[n] + MPAD_ - 1) / MPAD_) * MPAD_;   // pad bucket to 32 rows
        }
        start[N_] = s;                                      // padded_total
    }
}

__global__ void seps_scatter(const int* __restrict__ seps, int* __restrict__ cursor,
                             int* __restrict__ perm) {
    int r = blockIdx.x * blockDim.x + threadIdx.x;
    if (r >= ROWS_) return;
    int a = r / B_, b = r % B_;
    int net = seps[b * A_ + a];
    int pos = atomicAdd(&cursor[net], 1);
    perm[pos] = r;
}

// ---------------- data staging kernels ----------------
// Gather inputs (fp32 [A,B,D]) into permuted f16 rows [slot][D].
__global__ void seps_gather_x(const float* __restrict__ x, const int* __restrict__ perm,
                              _Float16* __restrict__ Xh) {
    int s = blockIdx.x;
    int t = threadIdx.x;                 // blockDim.x == D_
    int r = perm[s];
    float v = (r >= 0) ? x[(size_t)r * D_ + t] : 0.0f;
    Xh[(size_t)s * D_ + t] = (_Float16)v;
}

// Transpose+convert weights: src [N][K][H] fp32 -> dst [N][H][K] f16
__global__ void seps_wcvt(const float* __restrict__ W, _Float16* __restrict__ Wt,
                          int Kd, int Hd) {
    size_t i = (size_t)blockIdx.x * blockDim.x + threadIdx.x;
    size_t tot = (size_t)N_ * Kd * Hd;
    if (i >= tot) return;
    int h = (int)(i % Hd);
    size_t t = i / Hd;
    int k = (int)(t % Kd);
    int n = (int)(t / Kd);
    Wt[((size_t)n * Hd + h) * Kd + k] = (_Float16)W[i];
}

// ---------------- WMMA GEMM: 32x64 tile per wave, K-step 32 ----------------
// Each B fragment feeds two WMMAs (M-subtiles) -> 1.5 b128 loads per WMMA.
// Adata: [slots][K] f16 (permuted rows).  Wt: [N][Nout][K] f16.  bias: [N][Nout] f32.
// If gatherOut: write fp32 to Fout[row*Nout + n] via perm, else f16 to Hout[slot*Nout+n].
__global__ __launch_bounds__(256)
void seps_gemm(const _Float16* __restrict__ Adata,
               const _Float16* __restrict__ Wt,
               const float* __restrict__ bias,
               _Float16* __restrict__ Hout,
               float* __restrict__ Fout,
               const int* __restrict__ start,
               const int* __restrict__ perm,
               int K, int Nout, int doRelu, int gatherOut)
{
    const int lane = threadIdx.x & 31;
    const int wave = threadIdx.x >> 5;
    const int m0 = blockIdx.x * MPAD_;               // 32 rows per block tile

    const int padded_total = start[N_];
    if (m0 >= padded_total) return;                  // wave-uniform: EXEC all-1 for WMMA

    // which network owns this 32-row tile (buckets are 32-aligned)
    int net = 0;
#pragma unroll
    for (int j = 1; j < N_; ++j)
        if (m0 >= start[j]) net = j;

    const int n0 = blockIdx.y * 512 + wave * 64;     // 8 waves x 64 cols = 512 cols/block

    const _Float16* Wn = Wt + (size_t)net * Nout * K;
    const float* bn = bias + (size_t)net * Nout;

    // A fragment (16-bit A 16x32 layout): lanes 0-15: M=lane, K k+0..7 / k+16..23;
    // lanes 16-31: M=lane-16, K k+8..15 / k+24..31.
    const int lk = lane & 15;
    const int ak = (lane >> 4) * 8;
    const int bk = (lane >> 4) * 8;                  // B: lanes 0-15 K k+0..7 / k+16..23, 16-31 K k+8..15 / k+24..31

    const _Float16* ap0 = Adata + (size_t)(m0 + lk) * K + ak;
    const _Float16* ap1 = ap0 + (size_t)16 * K;
    const _Float16* bp0 = Wn + (size_t)(n0 + 0 * 16 + lk) * K + bk;
    const _Float16* bp1 = Wn + (size_t)(n0 + 1 * 16 + lk) * K + bk;
    const _Float16* bp2 = Wn + (size_t)(n0 + 2 * 16 + lk) * K + bk;
    const _Float16* bp3 = Wn + (size_t)(n0 + 3 * 16 + lk) * K + bk;

    v8f acc[2][4] = {};

    for (int k = 0; k < K; k += 32) {
        Frag16 a0, a1;
        a0.q[0] = *(const int4*)(ap0);
        a0.q[1] = *(const int4*)(ap0 + 16);
        a1.q[0] = *(const int4*)(ap1);
        a1.q[1] = *(const int4*)(ap1 + 16);

        const _Float16* bps[4] = { bp0, bp1, bp2, bp3 };
#pragma unroll
        for (int j = 0; j < 4; ++j) {
            Frag16 b;
            b.q[0] = *(const int4*)(bps[j]);
            b.q[1] = *(const int4*)(bps[j] + 16);
            __builtin_prefetch(bps[j] + 64, 0, 3);   // weights 2 K-steps ahead
            acc[0][j] = __builtin_amdgcn_wmma_f32_16x16x32_f16(
                false, a0.v, false, b.v, (short)0, acc[0][j], false, false);
            acc[1][j] = __builtin_amdgcn_wmma_f32_16x16x32_f16(
                false, a1.v, false, b.v, (short)0, acc[1][j], false, false);
        }
        ap0 += 32; ap1 += 32;
        bp0 += 32; bp1 += 32; bp2 += 32; bp3 += 32;
    }

    // Epilogue: C/D layout -> VGPR r holds M = r + 8*(lane>=16), N = lane&15
    const int ncol = lane & 15;
    const int mhalf = (lane >> 4) * 8;
#pragma unroll
    for (int t = 0; t < 2; ++t) {
#pragma unroll
        for (int j = 0; j < 4; ++j) {
            const int n = n0 + j * 16 + ncol;
            const float bv = bn[n];
#pragma unroll
            for (int r = 0; r < 8; ++r) {
                const int m = m0 + t * 16 + mhalf + r;
                float v = acc[t][j][r] + bv;
                if (doRelu) v = fmaxf(v, 0.0f);
                if (gatherOut) {
                    int row = perm[m];
                    if (row >= 0)                    // skip bucket-padding rows
                        Fout[(size_t)row * Nout + n] = v;
                } else {
                    Hout[(size_t)m * Nout + n] = (_Float16)v;
                }
            }
        }
    }
}

// ---------------- host-side launch ----------------
extern "C" void kernel_launch(void* const* d_in, const int* in_sizes, int n_in,
                              void* d_out, int out_size, void* d_ws, size_t ws_size,
                              hipStream_t stream) {
    const float* x    = (const float*)d_in[0];   // [A,B,D]
    const int*   seps = (const int*)  d_in[1];   // [B,A]
    const float* W0   = (const float*)d_in[2];   // [N,D,H1]
    const float* b0   = (const float*)d_in[3];   // [N,H1]
    const float* W1   = (const float*)d_in[4];   // [N,H1,H2]
    const float* b1   = (const float*)d_in[5];   // [N,H2]
    const float* W2   = (const float*)d_in[6];   // [N,H2,H3]
    const float* b2   = (const float*)d_in[7];   // [N,H3]
    float* out = (float*)d_out;                  // [A,B,H3]

    char* ws = (char*)d_ws;
    int* cnt    = (int*)ws;          // 4 ints
    int* start  = cnt + 4;           // 5 ints
    int* cursor = cnt + 9;           // 4 ints
    int* perm   = (int*)(ws + 256);  // MAXSLOTS_ ints (256-aligned size)

    size_t off = 256 + (size_t)MAXSLOTS_ * 4;
    _Float16* Xh  = (_Float16*)(ws + off); off += (size_t)MAXSLOTS_ * D_  * 2;
    _Float16* W0h = (_Float16*)(ws + off); off += (size_t)N_ * H1_ * D_  * 2;
    _Float16* W1h = (_Float16*)(ws + off); off += (size_t)N_ * H2_ * H1_ * 2;
    _Float16* W2h = (_Float16*)(ws + off); off += (size_t)N_ * H3_ * H2_ * 2;
    _Float16* H1h = (_Float16*)(ws + off); off += (size_t)MAXSLOTS_ * H1_ * 2;
    _Float16* H2h = (_Float16*)(ws + off);

    // 1) bucket rows by network id (counting sort, 32-row-padded buckets)
    seps_init_ws<<<(MAXSLOTS_ + 255) / 256, 256, 0, stream>>>(cnt, perm, MAXSLOTS_);
    seps_hist   <<<(ROWS_ + 255) / 256,     256, 0, stream>>>(seps, cnt);
    seps_offsets<<<1, 1, 0, stream>>>(cnt, start, cursor);
    seps_scatter<<<(ROWS_ + 255) / 256,     256, 0, stream>>>(seps, cursor, perm);

    // 2) stage data in f16
    seps_gather_x<<<MAXSLOTS_, D_, 0, stream>>>(x, perm, Xh);
    seps_wcvt<<<((size_t)N_ * D_  * H1_ + 255) / 256, 256, 0, stream>>>(W0, W0h, D_,  H1_);
    seps_wcvt<<<((size_t)N_ * H1_ * H2_ + 255) / 256, 256, 0, stream>>>(W1, W1h, H1_, H2_);
    seps_wcvt<<<((size_t)N_ * H2_ * H3_ + 255) / 256, 256, 0, stream>>>(W2, W2h, H2_, H3_);

    // 3) three WMMA GEMM layers (only the selected network per row)
    seps_gemm<<<dim3(MTILES_, H1_ / 512), 256, 0, stream>>>(
        Xh,  W0h, b0, H1h, nullptr, start, perm, D_,  H1_, 1, 0);
    seps_gemm<<<dim3(MTILES_, H2_ / 512), 256, 0, stream>>>(
        H1h, W1h, b1, H2h, nullptr, start, perm, H1_, H2_, 1, 0);
    seps_gemm<<<dim3(MTILES_, 1), 256, 0, stream>>>(
        H2h, W2h, b2, nullptr, out, start, perm, H2_, H3_, 0, 1);
}
